// Basenet_50225347560071
// MI455X (gfx1250) — compile-verified, matrix operand order checked by reading, not run on hardware
//
#include <hip/hip_runtime.h>
#include <hip/hip_bf16.h>

#define BT 16
#define D_CH 1056
#define OH 57
#define OW 87
#define MAXN 13
#define NFB 1024
#define NA 6
#define R_TOT 208            // BT*MAXN
#define K1 26400             // D_CH*25
#define NPAIR 156            // MAXN*(MAXN-1)
#define M2 2496              // BT*NPAIR
#define K2 2048              // 2*NFB

typedef __attribute__((ext_vector_type(16))) __bf16        v16bf;
typedef __attribute__((ext_vector_type(8)))  float         v8f;
typedef __attribute__((ext_vector_type(8)))  unsigned int  v8u;
typedef __attribute__((ext_vector_type(4)))  unsigned int  v4u;

// ---- f32 -> bf16 conversion helpers -------------------------------------
#if __has_builtin(__builtin_amdgcn_cvt_pk_bf16_f32)
__device__ __forceinline__ unsigned int pk_bf16(float lo, float hi) {
    return __builtin_bit_cast(unsigned int,
                              __builtin_amdgcn_cvt_pk_bf16_f32(lo, hi));
}
#else
__device__ __forceinline__ unsigned int pk_bf16(float lo, float hi) {
    unsigned int ulo = __builtin_bit_cast(unsigned int, lo);
    unsigned int uhi = __builtin_bit_cast(unsigned int, hi);
    ulo = (ulo + 0x7FFFu + ((ulo >> 16) & 1u)) >> 16;
    uhi = (uhi + 0x7FFFu + ((uhi >> 16) & 1u)) >> 16;
    return (uhi << 16) | (ulo & 0xFFFFu);
}
#endif

__device__ __forceinline__ unsigned short bf16_1(float f) {
    unsigned int u = __builtin_bit_cast(unsigned int, f);
    return (unsigned short)((u + 0x7FFFu + ((u >> 16) & 1u)) >> 16);
}

// ---- WMMA operand loaders (16-bit A/B lane layout, wave32) --------------
// Lane needs K = {kb..kb+7} and {16+kb..16+kb+7}  (kb = 0 or 8) -> two
// contiguous 16B chunks when the operand is bf16 in memory.
__device__ __forceinline__ v16bf load_bf16_operand(const unsigned short* base, int kb) {
    v4u lo = *(const v4u*)(base + kb);
    v4u hi = *(const v4u*)(base + 16 + kb);
    v8u r;
#pragma unroll
    for (int i = 0; i < 4; ++i) { r[i] = lo[i]; r[i + 4] = hi[i]; }
    return __builtin_bit_cast(v16bf, r);
}

// Fallback path: operand is f32 in memory; pack in-loop.
__device__ __forceinline__ v16bf load_f32_operand(const float* base, int kb) {
    v8u r;
#pragma unroll
    for (int i = 0; i < 8; ++i) {
        int off = (i < 4) ? (kb + 2 * i) : (16 + kb + 2 * (i - 4));
        float2 v = *(const float2*)(base + off);
        r[i] = pk_bf16(v.x, v.y);
    }
    return __builtin_bit_cast(v16bf, r);
}

// ---------------------------------------------------------------------------
// Stage 0: bulk f32 -> bf16 conversion (pairs)
// ---------------------------------------------------------------------------
__global__ void cvt_bf16_kernel(const float* __restrict__ src,
                                unsigned int* __restrict__ dst, int npairs) {
    int i = blockIdx.x * 256 + threadIdx.x;
    if (i >= npairs) return;
    float2 v = *(const float2*)(src + 2 * i);
    dst[i] = pk_bf16(v.x, v.y);
}

// ---------------------------------------------------------------------------
// Stage 1: ROI-align -> A1 bf16 [208][26400] row-major
// ---------------------------------------------------------------------------
__global__ void roi_kernel(const float* __restrict__ features,
                           const float* __restrict__ boxes,
                           unsigned short* __restrict__ A1) {
    int idx = blockIdx.x * 256 + threadIdx.x;
    if (idx >= R_TOT * D_CH * 25) return;
    int bin = idx % 25;
    int t   = idx / 25;
    int d   = t % D_CH;
    int r   = t / D_CH;
    int b   = r / MAXN;

    const float* bx = boxes + r * 4;
    float x1 = bx[0], y1 = bx[1], x2 = bx[2], y2 = bx[3];
    float bw = (x2 - x1) * 0.2f, bh = (y2 - y1) * 0.2f;
    int kx = bin % 5, ky = bin / 5;
    float xs = x1 + ((float)kx + 0.5f) * bw;
    float ys = y1 + ((float)ky + 0.5f) * bh;
    float xf = floorf(xs), yf = floorf(ys);
    float lx = xs - xf, ly = ys - yf;
    int x0 = (int)xf, y0 = (int)yf;
    int x0i = min(max(x0, 0), OW - 1), x1i = min(max(x0 + 1, 0), OW - 1);
    int y0i = min(max(y0, 0), OH - 1), y1i = min(max(y0 + 1, 0), OH - 1);

    const float* fp = features + (size_t)(b * D_CH + d) * (OH * OW);
    float v00 = fp[y0i * OW + x0i], v01 = fp[y0i * OW + x1i];
    float v10 = fp[y1i * OW + x0i], v11 = fp[y1i * OW + x1i];
    float w0 = 1.0f - ly, w1 = ly, u0 = 1.0f - lx, u1 = lx;
    float val = v00 * w0 * u0 + v01 * w0 * u1 + v10 * w1 * u0 + v11 * w1 * u1;
    A1[idx] = bf16_1(val);
}

// ---------------------------------------------------------------------------
// Stage 2: feats_raw = A1 @ W_emb^T + b_emb   (M=208, K=26400, N=1024)
// 8 waves/block, wave = 16x16 tile, block tile 16M x 128N.
// Variant A: W_emb pre-converted to bf16 (4x global_load_b128 + wmma / step)
// ---------------------------------------------------------------------------
__global__ void gemm1_bf16b_kernel(const unsigned short* __restrict__ A1,
                                   const unsigned short* __restrict__ Bbf,
                                   const float* __restrict__ b_emb,
                                   float* __restrict__ feats_raw) {
    int wave = threadIdx.x >> 5;
    int lane = threadIdx.x & 31;
    int l16  = lane & 15;
    int kb   = (lane >> 4) << 3;
    int mbase = blockIdx.x * 16;
    int n     = blockIdx.y * 128 + wave * 16 + l16;

    const unsigned short* aRow = A1  + (size_t)(mbase + l16) * K1;
    const unsigned short* bRow = Bbf + (size_t)n * K1;

    v8f c = {};
    for (int k0 = 0; k0 < K1; k0 += 32) {
        v16bf a = load_bf16_operand(aRow + k0, kb);
        v16bf b = load_bf16_operand(bRow + k0, kb);
        __builtin_prefetch(bRow + k0 + 512, 0, 1);   // global_prefetch_b8
        c = __builtin_amdgcn_wmma_f32_16x16x32_bf16(false, a, false, b,
                                                    (short)0, c, false, false);
    }

    float be  = b_emb[n];
    int   mhi = (lane >> 4) << 3;
#pragma unroll
    for (int r = 0; r < 8; ++r)
        feats_raw[(size_t)(mbase + mhi + r) * NFB + n] = c[r] + be;
}

// Variant B (small-ws fallback): W_emb streamed as f32, packed in-loop.
__global__ void gemm1_f32b_kernel(const unsigned short* __restrict__ A1,
                                  const float* __restrict__ W_emb,
                                  const float* __restrict__ b_emb,
                                  float* __restrict__ feats_raw) {
    int wave = threadIdx.x >> 5;
    int lane = threadIdx.x & 31;
    int l16  = lane & 15;
    int kb   = (lane >> 4) << 3;
    int mbase = blockIdx.x * 16;
    int n     = blockIdx.y * 128 + wave * 16 + l16;

    const unsigned short* aRow = A1    + (size_t)(mbase + l16) * K1;
    const float*          bRow = W_emb + (size_t)n * K1;

    v8f c = {};
    for (int k0 = 0; k0 < K1; k0 += 32) {
        v16bf a = load_bf16_operand(aRow + k0, kb);
        v16bf b = load_f32_operand(bRow + k0, kb);
        __builtin_prefetch(bRow + k0 + 512, 0, 1);
        c = __builtin_amdgcn_wmma_f32_16x16x32_bf16(false, a, false, b,
                                                    (short)0, c, false, false);
    }

    float be  = b_emb[n];
    int   mhi = (lane >> 4) << 3;
#pragma unroll
    for (int r = 0; r < 8; ++r)
        feats_raw[(size_t)(mbase + mhi + r) * NFB + n] = c[r] + be;
}

// ---------------------------------------------------------------------------
// Stage 3: LayerNorm + ReLU + actions head; writes feats_relu as bf16
// ---------------------------------------------------------------------------
__global__ void ln_act_kernel(const float* __restrict__ feats_raw,
                              const float* __restrict__ ln_g,
                              const float* __restrict__ ln_b,
                              const float* __restrict__ W_act,
                              const float* __restrict__ b_act,
                              unsigned short* __restrict__ feats_relu,
                              float* __restrict__ out_actions) {
    __shared__ float r1[256], r2[256];
    int row = blockIdx.x, tid = threadIdx.x;
    const float* x = feats_raw + (size_t)row * NFB;

    float v[4], s = 0.f, s2 = 0.f;
#pragma unroll
    for (int i = 0; i < 4; ++i) {
        v[i] = x[tid + 256 * i];
        s += v[i]; s2 += v[i] * v[i];
    }
    r1[tid] = s; r2[tid] = s2; __syncthreads();
    for (int st = 128; st > 0; st >>= 1) {
        if (tid < st) { r1[tid] += r1[tid + st]; r2[tid] += r2[tid + st]; }
        __syncthreads();
    }
    float mu   = r1[0] * (1.0f / NFB);
    float var  = r2[0] * (1.0f / NFB) - mu * mu;
    float rstd = rsqrtf(var + 1e-5f);

    float pa[NA] = {0.f, 0.f, 0.f, 0.f, 0.f, 0.f};
#pragma unroll
    for (int i = 0; i < 4; ++i) {
        int   nn = tid + 256 * i;
        float y  = (v[i] - mu) * rstd * ln_g[nn] + ln_b[nn];
        y = fmaxf(y, 0.0f);
        feats_relu[(size_t)row * NFB + nn] = bf16_1(y);
#pragma unroll
        for (int a = 0; a < NA; ++a) pa[a] += y * W_act[a * NFB + nn];
    }
    for (int a = 0; a < NA; ++a) {
        __syncthreads();
        r1[tid] = pa[a]; __syncthreads();
        for (int st = 128; st > 0; st >>= 1) {
            if (tid < st) r1[tid] += r1[tid + st];
            __syncthreads();
        }
        if (tid == 0) out_actions[row * NA + a] = r1[0] + b_act[a];
    }
}

// ---------------------------------------------------------------------------
// Stage 4: h = relu(pair @ W_i1^T + b_i1)  (M=2496, K=2048, N=1024)
// A rows gathered from feats_relu (bf16) via II/JJ; B = W_i1 bf16.
// ---------------------------------------------------------------------------
__global__ void gemm2_kernel(const unsigned short* __restrict__ feats_relu,
                             const unsigned short* __restrict__ Wi1bf,
                             const float* __restrict__ b_i1,
                             float* __restrict__ h) {
    int wave = threadIdx.x >> 5;
    int lane = threadIdx.x & 31;
    int l16  = lane & 15;
    int kb   = (lane >> 4) << 3;
    int mbase = blockIdx.x * 16;
    int n     = blockIdx.y * 128 + wave * 16 + l16;

    int pr = mbase + l16;               // pair row in [0, 2496)
    int bb = pr / NPAIR;
    int p  = pr % NPAIR;
    int ii = p / 12;
    int jm = p % 12;
    int jj = jm + (jm >= ii ? 1 : 0);

    const unsigned short* aI   = feats_relu + (size_t)(bb * MAXN + ii) * NFB;
    const unsigned short* aJ   = feats_relu + (size_t)(bb * MAXN + jj) * NFB;
    const unsigned short* bRow = Wi1bf + (size_t)n * K2;

    v8f c = {};
    for (int k0 = 0; k0 < K2; k0 += 32) {
        const unsigned short* aRow = (k0 < NFB) ? (aI + k0) : (aJ + (k0 - NFB));
        v16bf a = load_bf16_operand(aRow, kb);
        v16bf b = load_bf16_operand(bRow + k0, kb);
        c = __builtin_amdgcn_wmma_f32_16x16x32_bf16(false, a, false, b,
                                                    (short)0, c, false, false);
    }

    float bi  = b_i1[n];
    int   mhi = (lane >> 4) << 3;
#pragma unroll
    for (int r = 0; r < 8; ++r)
        h[(size_t)(mbase + mhi + r) * NFB + n] = fmaxf(c[r] + bi, 0.0f);
}

// ---------------------------------------------------------------------------
// Stage 5: interaction_scores = h @ W_i2^T + b_i2  ([2496,1024] -> [2496,2])
// ---------------------------------------------------------------------------
__global__ void inter_kernel(const float* __restrict__ h,
                             const float* __restrict__ W_i2,
                             const float* __restrict__ b_i2,
                             float* __restrict__ out_inter) {
    __shared__ float r1[256], r2[256];
    int row = blockIdx.x, tid = threadIdx.x;
    const float* hr = h + (size_t)row * NFB;
    float p0 = 0.f, p1 = 0.f;
#pragma unroll
    for (int i = 0; i < 4; ++i) {
        int   nn = tid + 256 * i;
        float y  = hr[nn];
        p0 += y * W_i2[nn];
        p1 += y * W_i2[NFB + nn];
    }
    r1[tid] = p0; r2[tid] = p1; __syncthreads();
    for (int st = 128; st > 0; st >>= 1) {
        if (tid < st) { r1[tid] += r1[tid + st]; r2[tid] += r2[tid + st]; }
        __syncthreads();
    }
    if (tid == 0) {
        out_inter[row * 2 + 0] = r1[0] + b_i2[0];
        out_inter[row * 2 + 1] = r2[0] + b_i2[1];
    }
}

// ---------------------------------------------------------------------------
extern "C" void kernel_launch(void* const* d_in, const int* in_sizes, int n_in,
                              void* d_out, int out_size, void* d_ws, size_t ws_size,
                              hipStream_t stream) {
    const float* features = (const float*)d_in[0];
    const float* boxes    = (const float*)d_in[1];
    // d_in[2] (bboxes_num_in) is constant MAXN in the reference -> unused
    const float* W_emb = (const float*)d_in[3];
    const float* b_emb = (const float*)d_in[4];
    const float* ln_g  = (const float*)d_in[5];
    const float* ln_b  = (const float*)d_in[6];
    const float* W_act = (const float*)d_in[7];
    const float* b_act = (const float*)d_in[8];
    const float* W_i1  = (const float*)d_in[9];
    const float* b_i1  = (const float*)d_in[10];
    const float* W_i2  = (const float*)d_in[11];
    const float* b_i2  = (const float*)d_in[12];

    float* out_actions = (float*)d_out;                 // [208*6]
    float* out_inter   = (float*)d_out + R_TOT * NA;    // [2496*2]

    // ---- workspace layout (256B-aligned sections) ----
    char* ws = (char*)d_ws;
    size_t off = 0;
    unsigned short* A1bf = (unsigned short*)(ws + off);      // 208*26400*2
    off += (size_t)R_TOT * K1 * 2;
    unsigned short* Wi1bf = (unsigned short*)(ws + off);     // 1024*2048*2
    off += (size_t)NFB * K2 * 2;
    float* feats_raw = (float*)(ws + off);                   // 208*1024*4
    off += (size_t)R_TOT * NFB * 4;
    unsigned short* feats_relu = (unsigned short*)(ws + off);// 208*1024*2
    off += (size_t)R_TOT * NFB * 2;
    float* h = (float*)(ws + off);                           // 2496*1024*4
    off += (size_t)M2 * NFB * 4;
    unsigned short* Wembbf = (unsigned short*)(ws + off);    // optional: 1024*26400*2
    size_t need_big = off + (size_t)NFB * K1 * 2;
    const bool big_ws = (ws_size >= need_big);               // deterministic per run

    // Stage 1: ROI align -> bf16 activation matrix
    {
        int n = R_TOT * D_CH * 25;
        roi_kernel<<<(n + 255) / 256, 256, 0, stream>>>(features, boxes, A1bf);
    }
    // Stage 0b: convert W_i1 (always) and W_emb (if ws allows) to bf16
    {
        int npairs = (NFB * K2) / 2;
        cvt_bf16_kernel<<<(npairs + 255) / 256, 256, 0, stream>>>(
            W_i1, (unsigned int*)Wi1bf, npairs);
    }
    if (big_ws) {
        int npairs = (NFB * K1) / 2;
        cvt_bf16_kernel<<<(npairs + 255) / 256, 256, 0, stream>>>(
            W_emb, (unsigned int*)Wembbf, npairs);
        gemm1_bf16b_kernel<<<dim3(R_TOT / 16, NFB / 128), 256, 0, stream>>>(
            A1bf, Wembbf, b_emb, feats_raw);
    } else {
        gemm1_f32b_kernel<<<dim3(R_TOT / 16, NFB / 128), 256, 0, stream>>>(
            A1bf, W_emb, b_emb, feats_raw);
    }
    // Stage 3: LayerNorm + ReLU + actions head
    ln_act_kernel<<<R_TOT, 256, 0, stream>>>(feats_raw, ln_g, ln_b, W_act, b_act,
                                             feats_relu, out_actions);
    // Stage 4: pair-interaction GEMM (bf16 WMMA, gathered A rows)
    gemm2_kernel<<<dim3(M2 / 16, NFB / 128), 256, 0, stream>>>(
        feats_relu, Wi1bf, b_i1, h);
    // Stage 5: interaction head
    inter_kernel<<<M2, 256, 0, stream>>>(h, W_i2, b_i2, out_inter);
}
